// TransformerLayerDecoder_89824946028699
// MI455X (gfx1250) — matrix operand order repaired and checked by reading, not compiled, not run on hardware
//
#include <hip/hip_runtime.h>
#include <hip/hip_bf16.h>
#include <math.h>

// ---------------------------------------------------------------------------
// Transformer decoder layer for MI455X (gfx1250, wave32, WMMA).
// T=1024, B=8, D=1024, H=16, NH=64.  fp32 in/out, f16 WMMA internally.
// ---------------------------------------------------------------------------

typedef __attribute__((ext_vector_type(16))) _Float16 v16h;
typedef __attribute__((ext_vector_type(8)))  _Float16 v8h;
typedef __attribute__((ext_vector_type(4)))  _Float16 v4h;
typedef __attribute__((ext_vector_type(8)))  float    v8f;

union V16 { v16h v; v8h h[2]; };

#define TT  1024
#define BB  8
#define DD  1024
#define HH  16
#define NHD 64

static __device__ __forceinline__ v8f wmma_f16(v16h a, v16h b, v8f c) {
  return __builtin_amdgcn_wmma_f32_16x16x32_f16(false, a, false, b, (short)0, c,
                                                false, false);
}

static __device__ __forceinline__ float grp16_max(float v) {
#pragma unroll
  for (int d = 1; d < 16; d <<= 1) v = fmaxf(v, __shfl_xor(v, d, 32));
  return v;
}
static __device__ __forceinline__ float grp16_sum(float v) {
#pragma unroll
  for (int d = 1; d < 16; d <<= 1) v += __shfl_xor(v, d, 32);
  return v;
}

// ---------------------------------------------------------------------------
// GEMM: C[M,1024] = A[M,1024] @ W[1024,1024] + bias
// Block = 256 thr (8 waves), block tile 128x256, wave tile 64x64 (4x4 WMMA),
// K-step 32, double-buffered LDS, WGP-scope global_prefetch of next k-tile.
// OUT_MODE 0: f16 row-major     (Q / K projections)
// OUT_MODE 1: f16, V-transposed: vt[((b*H+h)*64+c)*T + t]   (V projections)
// OUT_MODE 2: f32 row-major     (final linear)
// ---------------------------------------------------------------------------
template <int OUT_MODE>
__global__ __launch_bounds__(256) void gemm_kernel(
    const float* __restrict__ A, const float* __restrict__ W,
    const float* __restrict__ bias, void* __restrict__ Cout) {
  __shared__ _Float16 As[2][128][40];   // [m][k], row stride 80B (16B aligned)
  __shared__ _Float16 Bs[2][256][40];   // [n][k] (transposed W tile)

  const int tid  = threadIdx.x;
  const int lane = tid & 31;
  const int w    = tid >> 5;
  const int wm   = w & 1;    // 2 wave-rows (64 each)
  const int wn   = w >> 1;   // 4 wave-cols (64 each)
  const int bm   = blockIdx.y * 128;
  const int bn   = blockIdx.x * 256;
  const int lg   = lane >> 4;   // lane group 0/1
  const int lm   = lane & 15;

  v8f acc[4][4] = {};

  // --- staging: fp32 global -> f16 LDS (b128 global loads) ---
  const int s_ar = tid >> 3;          // A row base (0..31), step 32
  const int s_ac = (tid & 7) * 4;     // A col group (0..28)
  const int s_wk = tid >> 6;          // W k base (0..3), step 4
  const int s_wn = (tid & 63) * 4;    // W col group (0..252)

  auto stage = [&](int buf, int k0) {
#pragma unroll
    for (int i = 0; i < 4; i++) {
      int r = s_ar + i * 32;
      float4 va = *(const float4*)&A[(size_t)(bm + r) * DD + k0 + s_ac];
      v4h hv = {(_Float16)va.x, (_Float16)va.y, (_Float16)va.z, (_Float16)va.w};
      *(v4h*)&As[buf][r][s_ac] = hv;
    }
#pragma unroll
    for (int i = 0; i < 8; i++) {
      int kk = s_wk + i * 4;
      float4 vw = *(const float4*)&W[(size_t)(k0 + kk) * DD + bn + s_wn];
      Bs[buf][s_wn + 0][kk] = (_Float16)vw.x;
      Bs[buf][s_wn + 1][kk] = (_Float16)vw.y;
      Bs[buf][s_wn + 2][kk] = (_Float16)vw.z;
      Bs[buf][s_wn + 3][kk] = (_Float16)vw.w;
    }
  };

  auto compute = [&](int buf) {
    // A fragments: lane row m=lm; k halves at lg*8 and lg*8+16 (ISA A16x32)
    v16h af[4];
#pragma unroll
    for (int mt = 0; mt < 4; mt++) {
      const _Float16* ap = &As[buf][wm * 64 + mt * 16 + lm][0];
      V16 u;
      u.h[0] = *(const v8h*)(ap + lg * 8);
      u.h[1] = *(const v8h*)(ap + lg * 8 + 16);
      af[mt] = u.v;
    }
    // B fragments: lane col n=lm; k halves at lg*8 and lg*8+16 (matches A)
    v16h bf[4];
#pragma unroll
    for (int nt = 0; nt < 4; nt++) {
      const _Float16* bp = &Bs[buf][wn * 64 + nt * 16 + lm][0];
      V16 u;
      u.h[0] = *(const v8h*)(bp + lg * 8);
      u.h[1] = *(const v8h*)(bp + lg * 8 + 16);
      bf[nt] = u.v;
    }
#pragma unroll
    for (int mt = 0; mt < 4; mt++)
#pragma unroll
      for (int nt = 0; nt < 4; nt++)
        acc[mt][nt] = wmma_f16(af[mt], bf[nt], acc[mt][nt]);
  };

  stage(0, 0);
  __syncthreads();
  int buf = 0;
  for (int k0 = 32; k0 < DD; k0 += 32) {
    if (k0 + 32 < DD) {
      // WGP-scope prefetch (locality 3): pull next k-tile into all levels,
      // so the following stage()'s b128 loads hit the WGP cache.
      __builtin_prefetch(&A[(size_t)(bm + s_ar) * DD + k0 + 32 + s_ac], 0, 3);
      __builtin_prefetch(&W[(size_t)(k0 + 32 + s_wk) * DD + bn + s_wn], 0, 3);
    }
    stage(buf ^ 1, k0);
    compute(buf);
    __syncthreads();
    buf ^= 1;
  }
  compute(buf);

  // epilogue: C layout lane l, vgpr r -> (m = r + lg*8, n = lm)
#pragma unroll
  for (int mt = 0; mt < 4; mt++)
#pragma unroll
    for (int nt = 0; nt < 4; nt++)
#pragma unroll
      for (int r = 0; r < 8; r++) {
        int gm = bm + wm * 64 + mt * 16 + r + lg * 8;
        int gn = bn + wn * 64 + nt * 16 + lm;
        float v = acc[mt][nt][r] + bias[gn];
        if (OUT_MODE == 0) {
          ((_Float16*)Cout)[(size_t)gm * DD + gn] = (_Float16)v;
        } else if (OUT_MODE == 1) {
          int t = gm >> 3, b = gm & 7;               // row = t*B + b
          int vr = (b * HH + (gn >> 6)) * NHD + (gn & 63);
          ((_Float16*)Cout)[(size_t)vr * TT + t] = (_Float16)v;
        } else {
          ((float*)Cout)[(size_t)gm * DD + gn] = v;
        }
      }
}

// ---------------------------------------------------------------------------
// Flash attention: grid = (T/128, B*H), block = 256 (8 waves).
// Wave owns 16 query rows; streams 32-key tiles with online softmax.
// Q,K f16 row-major [T*B, D]; VT f16 transposed; O f32 row-major.
// ---------------------------------------------------------------------------
template <bool CAUSAL>
__global__ __launch_bounds__(256) void attn_kernel(
    const _Float16* __restrict__ Q, const _Float16* __restrict__ K,
    const _Float16* __restrict__ VT, float* __restrict__ O) {
  __shared__ _Float16 plds[8][16][32];  // per-wave P staging (C->A relayout)

  const int tid  = threadIdx.x;
  const int lane = tid & 31;
  const int w    = tid >> 5;
  const int lg   = lane >> 4;
  const int lm   = lane & 15;
  const int bh   = blockIdx.y;           // b*H + h
  const int b    = bh >> 4;
  const int h    = bh & 15;
  const int t0   = blockIdx.x * 128 + w * 16;

  // Q A-fragments (2 chunks over NH=64)
  v16h qa[2];
#pragma unroll
  for (int c = 0; c < 2; c++) {
    const _Float16* qp =
        Q + (size_t)((t0 + lm) * BB + b) * DD + h * NHD + c * 32;
    V16 u;
    u.h[0] = *(const v8h*)(qp + lg * 8);
    u.h[1] = *(const v8h*)(qp + lg * 8 + 16);
    qa[c] = u.v;
  }

  float mrow[8], srow[8];
#pragma unroll
  for (int r = 0; r < 8; r++) { mrow[r] = -INFINITY; srow[r] = 0.f; }
  v8f o[4] = {};

  const int jend = CAUSAL ? ((t0 + 15) >> 5) + 1 : (TT >> 5);
  for (int j = 0; j < jend; j++) {
    if (j + 1 < jend) {  // WGP-scope prefetch of next key/value tiles
      int ktn = (j + 1) * 32 + lm;
      __builtin_prefetch(K + (size_t)(ktn * BB + b) * DD + h * NHD + lg * 16,
                         0, 3);
      __builtin_prefetch(VT + (size_t)(bh * NHD + lm) * TT + (j + 1) * 32 +
                             lg * 16,
                         0, 3);
    }
    // K B-fragments: lane holds key n; nh contiguous in memory
    v16h kb[2][2];
#pragma unroll
    for (int c = 0; c < 2; c++)
#pragma unroll
      for (int nh = 0; nh < 2; nh++) {
        int kt = j * 32 + nh * 16 + lm;
        const _Float16* kp =
            K + (size_t)(kt * BB + b) * DD + h * NHD + c * 32 + lg * 8;
        V16 u;
        u.h[0] = *(const v8h*)kp;
        u.h[1] = *(const v8h*)(kp + 16);
        kb[c][nh] = u.v;
      }
    v8f s0 = {}, s1 = {};
    s0 = wmma_f16(qa[0], kb[0][0], s0);
    s0 = wmma_f16(qa[1], kb[1][0], s0);
    s1 = wmma_f16(qa[0], kb[0][1], s1);
    s1 = wmma_f16(qa[1], kb[1][1], s1);

#pragma unroll
    for (int r = 0; r < 8; r++) {
      float a = s0[r] * 0.125f;
      float c2 = s1[r] * 0.125f;
      if (CAUSAL) {
        int qt  = t0 + r + lg * 8;
        int k0i = j * 32 + lm;
        if (k0i > qt) a = -INFINITY;
        if (k0i + 16 > qt) c2 = -INFINITY;
      }
      float tm = grp16_max(fmaxf(a, c2));
      float nm = fmaxf(mrow[r], tm);
      float corr = __expf(mrow[r] - nm);
      float p0 = __expf(a - nm);
      float p1 = __expf(c2 - nm);
      float rs = grp16_sum(p0 + p1);
      srow[r] = srow[r] * corr + rs;
      mrow[r] = nm;
      o[0][r] *= corr; o[1][r] *= corr; o[2][r] *= corr; o[3][r] *= corr;
      plds[w][r + lg * 8][lm]      = (_Float16)p0;
      plds[w][r + lg * 8][lm + 16] = (_Float16)p1;
    }
    // reload P as A-fragment (wave-private LDS; compiler inserts dscnt waits)
    V16 pu;
    const _Float16* pp = &plds[w][lm][0];
    pu.h[0] = *(const v8h*)(pp + lg * 8);
    pu.h[1] = *(const v8h*)(pp + lg * 8 + 16);
    v16h pa = pu.v;
    // V B-fragments from transposed layout: contiguous over keys
#pragma unroll
    for (int c = 0; c < 4; c++) {
      const _Float16* vp =
          VT + (size_t)(bh * NHD + c * 16 + lm) * TT + j * 32 + lg * 8;
      V16 u;
      u.h[0] = *(const v8h*)vp;
      u.h[1] = *(const v8h*)(vp + 16);
      o[c] = wmma_f16(pa, u.v, o[c]);
    }
  }

#pragma unroll
  for (int c = 0; c < 4; c++)
#pragma unroll
    for (int r = 0; r < 8; r++) {
      int t = t0 + r + lg * 8;
      float val = o[c][r] / srow[r];
      O[(size_t)(t * BB + b) * DD + h * NHD + c * 16 + lm] = val;
    }
}

// ---------------------------------------------------------------------------
// Out[row,:] = LayerNorm(Aa[row,:] + Bb[row,:]) * g + be  (one block per row)
// Safe when Out aliases Bb (element read before write, per-thread).
// ---------------------------------------------------------------------------
__global__ __launch_bounds__(256) void ln_residual_kernel(
    const float* __restrict__ Aa, const float* __restrict__ Bb,
    const float* __restrict__ g, const float* __restrict__ be,
    float* __restrict__ Out) {
  __shared__ float red[256];
  const int row = blockIdx.x;
  const int tid = threadIdx.x;
  float v[4];
  float sum = 0.f;
#pragma unroll
  for (int i = 0; i < 4; i++) {
    int c = tid + i * 256;
    v[i] = Aa[(size_t)row * DD + c] + Bb[(size_t)row * DD + c];
    sum += v[i];
  }
  red[tid] = sum;
  __syncthreads();
  for (int s = 128; s > 0; s >>= 1) {
    if (tid < s) red[tid] += red[tid + s];
    __syncthreads();
  }
  float mean = red[0] * (1.0f / DD);
  __syncthreads();
  float sq = 0.f;
#pragma unroll
  for (int i = 0; i < 4; i++) {
    float d = v[i] - mean;
    sq += d * d;
  }
  red[tid] = sq;
  __syncthreads();
  for (int s = 128; s > 0; s >>= 1) {
    if (tid < s) red[tid] += red[tid + s];
    __syncthreads();
  }
  float inv = rsqrtf(red[0] * (1.0f / DD) + 1e-5f);
#pragma unroll
  for (int i = 0; i < 4; i++) {
    int c = tid + i * 256;
    Out[(size_t)row * DD + c] = (v[i] - mean) * inv * g[c] + be[c];
  }
}

// ---------------------------------------------------------------------------
extern "C" void kernel_launch(void* const* d_in, const int* in_sizes, int n_in,
                              void* d_out, int out_size, void* d_ws,
                              size_t ws_size, hipStream_t stream) {
  const float* enc = (const float*)d_in[0];
  const float* dec = (const float*)d_in[1];
  const float* Wq1 = (const float*)d_in[2];
  const float* bq1 = (const float*)d_in[3];
  const float* Wk1 = (const float*)d_in[4];
  const float* bk1 = (const float*)d_in[5];
  const float* Wv1 = (const float*)d_in[6];
  const float* bv1 = (const float*)d_in[7];
  const float* Wq2 = (const float*)d_in[8];
  const float* bq2 = (const float*)d_in[9];
  const float* Wk2 = (const float*)d_in[10];
  const float* bk2 = (const float*)d_in[11];
  const float* Wv2 = (const float*)d_in[12];
  const float* bv2 = (const float*)d_in[13];
  const float* Wl  = (const float*)d_in[14];
  const float* bl  = (const float*)d_in[15];
  const float* g1  = (const float*)d_in[16];
  const float* be1 = (const float*)d_in[17];
  const float* g2  = (const float*)d_in[18];
  const float* be2 = (const float*)d_in[19];
  const float* g3  = (const float*)d_in[20];
  const float* be3 = (const float*)d_in[21];

  char* ws = (char*)d_ws;
  _Float16* qb   = (_Float16*)(ws + (size_t)0);          // 16 MiB
  _Float16* kb   = (_Float16*)(ws + ((size_t)16 << 20)); // 16 MiB
  _Float16* vt   = (_Float16*)(ws + ((size_t)32 << 20)); // 16 MiB
  float*    attn = (float*)   (ws + ((size_t)48 << 20)); // 32 MiB
  float*    xbuf = (float*)   (ws + ((size_t)80 << 20)); // 32 MiB (x1 -> y1)
  float*    zbuf = (float*)   (ws + (size_t)0);          // reuse q+k (32 MiB)
  float*    out  = (float*)d_out;

  const dim3 gg(DD / 256, (TT * BB) / 128);  // (4, 64)
  const dim3 ga(TT / 128, BB * HH);          // (8, 128)
  const int NROW = TT * BB;                  // 8192

  // ---- self-attention block (causal) ----
  gemm_kernel<0><<<gg, 256, 0, stream>>>(dec, Wq1, bq1, qb);
  gemm_kernel<0><<<gg, 256, 0, stream>>>(dec, Wk1, bk1, kb);
  gemm_kernel<1><<<gg, 256, 0, stream>>>(dec, Wv1, bv1, vt);
  attn_kernel<true><<<ga, 256, 0, stream>>>(qb, kb, vt, attn);
  ln_residual_kernel<<<NROW, 256, 0, stream>>>(attn, dec, g1, be1, xbuf);

  // ---- cross-attention block ----
  gemm_kernel<0><<<gg, 256, 0, stream>>>(xbuf, Wq2, bq2, qb);
  gemm_kernel<0><<<gg, 256, 0, stream>>>(enc, Wk2, bk2, kb);
  gemm_kernel<1><<<gg, 256, 0, stream>>>(enc, Wv2, bv2, vt);
  attn_kernel<false><<<ga, 256, 0, stream>>>(qb, kb, vt, attn);
  ln_residual_kernel<<<NROW, 256, 0, stream>>>(attn, xbuf, g2, be2, xbuf);

  // ---- position-wise linear + residual + LN ----
  gemm_kernel<2><<<gg, 256, 0, stream>>>(xbuf, Wl, bl, zbuf);
  ln_residual_kernel<<<NROW, 256, 0, stream>>>(zbuf, xbuf, g3, be3, out);
}
